// PointNetSetAbstraction_9259949491067
// MI455X (gfx1250) — matrix operand order repaired and checked by reading, not compiled, not run on hardware
//
#include <hip/hip_runtime.h>

// Problem constants (match reference)
#define NPTS   4096
#define BATCH  8
#define SPTS   1024    // NPOINT
#define KSAMP  32
#define CIN    64
#define R2     (0.2f * 0.2f)
#define EPS_BN 1e-5f

typedef _Float16 v16h __attribute__((ext_vector_type(16)));
typedef _Float16 v8h  __attribute__((ext_vector_type(8)));
typedef float    v8f  __attribute__((ext_vector_type(8)));

#define WMMA(A, B, C) \
  __builtin_amdgcn_wmma_f32_16x16x32_f16(false, (A), false, (B), (short)0, (C), false, false)

// ---------------------------------------------------------------------------
// Kernel 1: Farthest point sampling, one block per batch.
// ---------------------------------------------------------------------------
__global__ __launch_bounds__(1024) void fps_kernel(const float* __restrict__ xyz,
                                                   float* __restrict__ newxyz) {
  __shared__ float lx[NPTS], ly[NPTS], lz[NPTS];
  __shared__ int   idxbuf[SPTS];
  __shared__ float wbv[32];
  __shared__ int   wbi[32];
  __shared__ int   sFar;

  const int b = blockIdx.x, tid = threadIdx.x;
  const int lane = tid & 31, wid = tid >> 5;

  for (int i = tid; i < NPTS; i += 1024) {
    const float* p = &xyz[(b * NPTS + i) * 3];
    lx[i] = p[0]; ly[i] = p[1]; lz[i] = p[2];
  }
  __syncthreads();

  float dmin[4] = {1e10f, 1e10f, 1e10f, 1e10f};
  int far = 0;

  for (int it = 0; it < SPTS; ++it) {
    if (tid == 0) idxbuf[it] = far;
    const float cx = lx[far], cy = ly[far], cz = lz[far];
    float bv = -1.0f; int bi = 0;
#pragma unroll
    for (int j = 0; j < 4; ++j) {
      const int p = tid + 1024 * j;                 // ascending index order
      const float dx = lx[p] - cx, dy = ly[p] - cy, dz = lz[p] - cz;
      const float d = dx * dx + dy * dy + dz * dz;
      dmin[j] = fminf(dmin[j], d);
      if (dmin[j] > bv) { bv = dmin[j]; bi = p; }   // strict > keeps first on ties
    }
    for (int off = 16; off; off >>= 1) {            // wave32 argmax reduce
      const float ov = __shfl_xor(bv, off);
      const int   oi = __shfl_xor(bi, off);
      if (ov > bv || (ov == bv && oi < bi)) { bv = ov; bi = oi; }
    }
    if (lane == 0) { wbv[wid] = bv; wbi[wid] = bi; }
    __syncthreads();
    if (wid == 0) {
      bv = wbv[lane]; bi = wbi[lane];
      for (int off = 16; off; off >>= 1) {
        const float ov = __shfl_xor(bv, off);
        const int   oi = __shfl_xor(bi, off);
        if (ov > bv || (ov == bv && oi < bi)) { bv = ov; bi = oi; }
      }
      if (lane == 0) sFar = bi;
    }
    __syncthreads();
    far = sFar;
  }

  if (tid < SPTS) {                                 // gather new_xyz (B,S,3)
    const int p = idxbuf[tid];
    float* o = &newxyz[(b * SPTS + tid) * 3];
    o[0] = lx[p]; o[1] = ly[p]; o[2] = lz[p];
  }
}

// ---------------------------------------------------------------------------
// Kernel 2: Ball query, one wave per centroid; ballot-ordered compaction.
// ---------------------------------------------------------------------------
__global__ __launch_bounds__(256) void ball_query_kernel(const float* __restrict__ xyz,
                                                         const float* __restrict__ newxyz,
                                                         int* __restrict__ gidx) {
  const int tid = threadIdx.x, lane = tid & 31, wv = tid >> 5;
  const int g = blockIdx.x * 8 + wv;                // centroid id in [0, B*S)
  const int b = g >> 10;
  const float qx = newxyz[g * 3 + 0], qy = newxyz[g * 3 + 1], qz = newxyz[g * 3 + 2];

  int cnt = 0, first = -1;
  for (int base = 0; base < NPTS && cnt < KSAMP; base += 32) {
    const int j = base + lane;
    const float* p = &xyz[(b * NPTS + j) * 3];
    const float dx = p[0] - qx, dy = p[1] - qy, dz = p[2] - qz;
    const bool pred = (dx * dx + dy * dy + dz * dz) <= R2;
    const unsigned m = (unsigned)__ballot(pred);
    if (first < 0 && m) first = base + __ffs(m) - 1;
    const int pos = cnt + __popc(m & ((1u << lane) - 1u));
    if (pred && pos < KSAMP) gidx[g * KSAMP + pos] = j;
    cnt += __popc(m);
  }
  if (cnt > KSAMP) cnt = KSAMP;
  for (int p2 = cnt + lane; p2 < KSAMP; p2 += 32) gidx[g * KSAMP + p2] = first;
}

// ---------------------------------------------------------------------------
// Kernel 3: gather + 3-layer MLP (WMMA f16 -> f32 accum) + BN/ReLU + K-max.
// 256 threads = 8 waves; 4 centroids (128 rows); wave owns a 16-row tile.
// A fragments: 2x ds_load_b128 per chunk (aligned, zero-padded, no guards).
// B fragments: weights pre-swizzled in LDS to fragment order -> 1 vector load.
// ---------------------------------------------------------------------------

#define STG_LD 96   // stg row stride in halves (192 B, 16-B aligned)

// A fragment: lane (rr,hf): elems 0..7 = K c0+8hf.., elems 8..15 = K c0+16+8hf..
__device__ __forceinline__ v16h load_a_frag(const _Float16* S, int c0, int lane) {
  const int rr = lane & 15, hf = lane >> 4;
  const v8h lo = *(const v8h*)&S[rr * STG_LD + c0 + 8 * hf];
  const v8h hi = *(const v8h*)&S[rr * STG_LD + c0 + 16 + 8 * hf];
  return __builtin_shufflevector(lo, hi, 0, 1, 2, 3, 4, 5, 6, 7,
                                 8, 9, 10, 11, 12, 13, 14, 15);
}

// B fragment from pre-swizzled LDS: [frag][lane][16] contiguous halves.
__device__ __forceinline__ v16h load_b_frag(const _Float16* swB, int frag, int lane) {
  return *(const v16h*)&swB[(frag * 32 + lane) * 16];
}

__global__ __launch_bounds__(256) void group_mlp_kernel(
    const float* __restrict__ xyz, const float* __restrict__ pts,
    const float* __restrict__ newxyz, const int* __restrict__ gidx,
    const float* __restrict__ w0, const float* __restrict__ g0, const float* __restrict__ b0,
    const float* __restrict__ m0, const float* __restrict__ v0,
    const float* __restrict__ w1, const float* __restrict__ g1, const float* __restrict__ b1,
    const float* __restrict__ m1, const float* __restrict__ v1,
    const float* __restrict__ w2, const float* __restrict__ g2, const float* __restrict__ b2,
    const float* __restrict__ m2, const float* __restrict__ v2,
    float* __restrict__ pooled) {
  // Weight fragments, swizzled: [nt*C + c][lane 32][j 16]
  __shared__ _Float16 sw0[12 * 512];   // 4 nt * 3 chunks  (12,288 B)
  __shared__ _Float16 sw1[8 * 512];    // 4 nt * 2 chunks  ( 8,192 B)
  __shared__ _Float16 sw2[16 * 512];   // 8 nt * 2 chunks  (16,384 B)
  __shared__ float    ssc[256], sbi[256];           // folded BN (2,048 B)
  __shared__ _Float16 stg[8 * 16 * STG_LD];         // activations (24,576 B)

  const int tid = threadIdx.x, lane = tid & 31, wv = tid >> 5;
  const int hf = lane >> 4, cl0 = lane & 15;

  // ---- swizzle weights into fragment order (f16), fold BN params ----
  for (int i = tid; i < 12 * 512; i += 256) {       // layer 0: K pad 67 -> 96
    const int grp = i >> 9, nt = grp / 3, c = grp % 3;
    const int ln = (i >> 4) & 31, j = i & 15;
    const int k = c * 32 + 8 * (ln >> 4) + (j & 7) + 16 * (j >> 3);
    const int col = nt * 16 + (ln & 15);
    sw0[i] = (k < 67) ? (_Float16)w0[col * 67 + k] : (_Float16)0.0f;
  }
  for (int i = tid; i < 8 * 512; i += 256) {        // layer 1
    const int grp = i >> 9, nt = grp >> 1, c = grp & 1;
    const int ln = (i >> 4) & 31, j = i & 15;
    const int k = c * 32 + 8 * (ln >> 4) + (j & 7) + 16 * (j >> 3);
    const int col = nt * 16 + (ln & 15);
    sw1[i] = (_Float16)w1[col * 64 + k];
  }
  for (int i = tid; i < 16 * 512; i += 256) {       // layer 2
    const int grp = i >> 9, nt = grp >> 1, c = grp & 1;
    const int ln = (i >> 4) & 31, j = i & 15;
    const int k = c * 32 + 8 * (ln >> 4) + (j & 7) + 16 * (j >> 3);
    const int col = nt * 16 + (ln & 15);
    sw2[i] = (_Float16)w2[col * 64 + k];
  }
  if (tid < 64) {
    const float s = g0[tid] * rsqrtf(v0[tid] + EPS_BN);
    ssc[tid] = s; sbi[tid] = b0[tid] - m0[tid] * s;
  } else if (tid < 128) {
    const int c = tid - 64;
    const float s = g1[c] * rsqrtf(v1[c] + EPS_BN);
    ssc[tid] = s; sbi[tid] = b1[c] - m1[c] * s;
  }
  if (tid < 128) {
    const float s = g2[tid] * rsqrtf(v2[tid] + EPS_BN);
    ssc[128 + tid] = s; sbi[128 + tid] = b2[tid] - m2[tid] * s;
  }

  // ---- gather: 2 threads per row, aligned v8h LDS stores ----
  const int gBase = blockIdx.x * 4;
  {
    const int row = tid >> 1, sub = tid & 1;        // 128 rows x 2
    const int g = gBase + (row >> 5), b = g >> 10;
    const int p = gidx[g * KSAMP + (row & 31)];
    const float* prow = &pts[(size_t)(b * CIN) * NPTS + p];   // +c*NPTS per channel
    _Float16* dst = &stg[row * STG_LD];
#pragma unroll
    for (int blk = 0; blk < 6; ++blk) {
      const int base = sub * 48 + blk * 8;
      v8h t;
#pragma unroll
      for (int j = 0; j < 8; ++j) {
        const int ch = base + j;
        float val;
        if (ch < 3)       val = xyz[(b * NPTS + p) * 3 + ch] - newxyz[g * 3 + ch];
        else if (ch < 67) val = prow[(ch - 3) * NPTS];
        else              val = 0.0f;
        t[j] = (_Float16)val;
      }
      *(v8h*)&dst[base] = t;
    }
  }
  __syncthreads();

  _Float16* S = &stg[wv * 16 * STG_LD];

  // ---- layer 1: 67(pad 96) -> 64   (3 K-chunks x 4 N-tiles = 12 WMMA) ----
  {
    const v16h a0 = load_a_frag(S, 0, lane);
    const v16h a1 = load_a_frag(S, 32, lane);
    const v16h a2 = load_a_frag(S, 64, lane);
#pragma unroll
    for (int nt = 0; nt < 4; ++nt) {
      v8f acc = {};
      acc = WMMA(a0, load_b_frag(sw0, nt * 3 + 0, lane), acc);
      acc = WMMA(a1, load_b_frag(sw0, nt * 3 + 1, lane), acc);
      acc = WMMA(a2, load_b_frag(sw0, nt * 3 + 2, lane), acc);
      const int ch = nt * 16 + cl0;
      const float sc = ssc[ch], bb = sbi[ch];
#pragma unroll
      for (int r = 0; r < 8; ++r) {
        const float y = fmaxf(acc[r] * sc + bb, 0.0f);
        S[(r + 8 * hf) * STG_LD + ch] = (_Float16)y;  // D elem r -> row r+8hf
      }
    }
  }

  // ---- layer 2: 64 -> 64   (2 x 4 = 8 WMMA) ----
  {
    const v16h a0 = load_a_frag(S, 0, lane);
    const v16h a1 = load_a_frag(S, 32, lane);
#pragma unroll
    for (int nt = 0; nt < 4; ++nt) {
      v8f acc = {};
      acc = WMMA(a0, load_b_frag(sw1, nt * 2 + 0, lane), acc);
      acc = WMMA(a1, load_b_frag(sw1, nt * 2 + 1, lane), acc);
      const int ch = nt * 16 + cl0;
      const float sc = ssc[64 + ch], bb = sbi[64 + ch];
#pragma unroll
      for (int r = 0; r < 8; ++r) {
        const float y = fmaxf(acc[r] * sc + bb, 0.0f);
        S[(r + 8 * hf) * STG_LD + ch] = (_Float16)y;
      }
    }
  }

  // ---- layer 3: 64 -> 128  (2 x 8 = 16 WMMA) + per-wave 16-row max ----
  {
    const v16h a0 = load_a_frag(S, 0, lane);
    const v16h a1 = load_a_frag(S, 32, lane);
    __syncthreads();   // fence: f16 stg reads done; slice reused as f32 pmax
    float* pm = (float*)S;                          // 8 nt x 32 lanes = 1 KB
#pragma unroll
    for (int nt = 0; nt < 8; ++nt) {
      v8f acc = {};
      acc = WMMA(a0, load_b_frag(sw2, nt * 2 + 0, lane), acc);
      acc = WMMA(a1, load_b_frag(sw2, nt * 2 + 1, lane), acc);
      const int ch = nt * 16 + cl0;
      const float sc = ssc[128 + ch], bb = sbi[128 + ch];
      float mx = 0.0f;                              // ReLU folded into max(0, .)
#pragma unroll
      for (int r = 0; r < 8; ++r) mx = fmaxf(mx, acc[r] * sc + bb);
      pm[nt * 32 + lane] = mx;                      // rows r+8hf covered per lane
    }
  }
  __syncthreads();

  // ---- combine wave partials -> pooled (B,128,S) ----
  for (int e = tid; e < 4 * 128; e += 256) {
    const int cl = e >> 7, ch = e & 127, nt = ch >> 4, l = ch & 15;
    const float* pA = (const float*)&stg[(cl * 2) * 16 * STG_LD];      // rows 0..15
    const float* pB = (const float*)&stg[(cl * 2 + 1) * 16 * STG_LD];  // rows 16..31
    const float v = fmaxf(fmaxf(pA[nt * 32 + l], pA[nt * 32 + l + 16]),
                          fmaxf(pB[nt * 32 + l], pB[nt * 32 + l + 16]));
    const int g = gBase + cl;
    pooled[((g >> 10) * 128 + ch) * SPTS + (g & 1023)] = v;
  }
}

// ---------------------------------------------------------------------------
extern "C" void kernel_launch(void* const* d_in, const int* in_sizes, int n_in,
                              void* d_out, int out_size, void* d_ws, size_t ws_size,
                              hipStream_t stream) {
  (void)in_sizes; (void)n_in; (void)out_size; (void)ws_size;
  const float* xyz = (const float*)d_in[0];
  const float* pts = (const float*)d_in[1];
  const float* w0 = (const float*)d_in[2];
  const float* g0 = (const float*)d_in[3];
  const float* b0 = (const float*)d_in[4];
  const float* m0 = (const float*)d_in[5];
  const float* v0 = (const float*)d_in[6];
  const float* w1 = (const float*)d_in[7];
  const float* g1 = (const float*)d_in[8];
  const float* b1 = (const float*)d_in[9];
  const float* m1 = (const float*)d_in[10];
  const float* v1 = (const float*)d_in[11];
  const float* w2 = (const float*)d_in[12];
  const float* g2 = (const float*)d_in[13];
  const float* b2 = (const float*)d_in[14];
  const float* m2 = (const float*)d_in[15];
  const float* v2 = (const float*)d_in[16];

  float* out = (float*)d_out;
  float* newxyz = out;                          // (B,S,3)
  float* pooled = out + BATCH * SPTS * 3;       // (B,128,S)
  int* gidx = (int*)d_ws;                       // (B,S,K) ints = 1 MB scratch

  fps_kernel<<<BATCH, 1024, 0, stream>>>(xyz, newxyz);
  ball_query_kernel<<<(BATCH * SPTS) / 8, 256, 0, stream>>>(xyz, newxyz, gidx);
  group_mlp_kernel<<<(BATCH * SPTS) / 4, 256, 0, stream>>>(
      xyz, pts, newxyz, gidx,
      w0, g0, b0, m0, v0, w1, g1, b1, m1, v1, w2, g2, b2, m2, v2, pooled);
}
